// KeypointDetection_8306466750937
// MI455X (gfx1250) — matrix operand order repaired and checked by reading, not compiled, not run on hardware
//
#include <hip/hip_runtime.h>
#include <hip/hip_bf16.h>
#include <stdint.h>

// ---------------------------------------------------------------------------
// CDNA5 (gfx1250) bf16 WMMA: V_WMMA_F32_16X16X32_BF16, wave32.
// A 16x32: lane m=lane&15 holds two contiguous 8-elem K runs
//          (K = kk + (lane<16?0:8) + 0..7  and  K = kk+16 + (lane<16?0:8) + 0..7)
// B 32x16: lane n=lane&15 holds two contiguous 8-elem K runs, same K order as A
//          (K = kk + (lane<16?0:8) + 0..7  and  K = kk+16 + (lane<16?0:8) + 0..7)  when weights are stored [n][k]
// C/D 16x16 f32: reg r, lane l -> (m = r + (l<16?0:8), n = l&15)
// ---------------------------------------------------------------------------
typedef __bf16 bf16;
typedef bf16  v8bf  __attribute__((ext_vector_type(8)));
typedef bf16  v16bf __attribute__((ext_vector_type(16)));
typedef float v8f   __attribute__((ext_vector_type(8)));

__device__ __forceinline__ v8f wmma_bf16(v16bf a, v16bf b, v8f c) {
  return __builtin_amdgcn_wmma_f32_16x16x32_bf16(false, a, false, b, (short)0, c,
                                                 false, false);
}

// A fragment: src row-major [16 x ld] bf16 tile (16B-aligned rows), K window kk.
__device__ __forceinline__ v16bf loadA(const bf16* __restrict__ src, int ld,
                                       int lane, int kk) {
  const int m = lane & 15;
  const int koff = (lane < 16) ? 0 : 8;
  const bf16* p = src + (size_t)m * ld + kk + koff;
  v8bf lo = *(const v8bf*)p;
  v8bf hi = *(const v8bf*)(p + 16);
  return __builtin_shufflevector(lo, hi, 0, 1, 2, 3, 4, 5, 6, 7,
                                 8, 9, 10, 11, 12, 13, 14, 15);
}

// B fragment from K-contiguous ("[n][k]") weights, row stride ld, N tile n0.
__device__ __forceinline__ v16bf loadBT(const bf16* __restrict__ wt, int ld,
                                        int lane, int n0, int kk) {
  const int n = n0 + (lane & 15);
  const bf16* p = wt + (size_t)n * ld + kk + ((lane < 16) ? 0 : 8);
  v8bf lo = *(const v8bf*)p;
  v8bf hi = *(const v8bf*)(p + 16);
  return __builtin_shufflevector(lo, hi, 0, 1, 2, 3, 4, 5, 6, 7,
                                 8, 9, 10, 11, 12, 13, 14, 15);
}

// Async global -> LDS staging (ASYNCcnt path), 16B per enabled lane.
__device__ __forceinline__ void async_copy16(const void* g, uint32_t lds_off) {
  asm volatile("global_load_async_to_lds_b128 %0, %1, off"
               :: "v"(lds_off), "v"((uint64_t)(uintptr_t)g) : "memory");
}
__device__ __forceinline__ void wait_async0() {
  asm volatile("s_wait_asynccnt 0" ::: "memory");
}

__device__ __forceinline__ float silu_f(float v) { return v / (1.f + __expf(-v)); }
__device__ __forceinline__ float sigmoid_f(float v) { return 1.f / (1.f + __expf(-v)); }

#define C_CH 32
#define KPN  17
#define MAXI 100
#define NCH  256
#define KPAR 2737

// ---------------------------------------------------------------------------
// Pack kernels (run once; weights amortize over 25..1024 consumer blocks)
// ---------------------------------------------------------------------------
__global__ void k_cvt(const float* __restrict__ in, bf16* __restrict__ out, int n) {
  const int i = blockIdx.x * 256 + threadIdx.x;
  if (i < n) out[i] = (bf16)in[i];
}
// out[b][j][i] = in[b][i][j]   (f32 -> bf16 transpose)
__global__ void k_tr(const float* __restrict__ in, bf16* __restrict__ out,
                     int B, int I, int J) {
  const size_t total = (size_t)B * I * J;
  for (size_t t = (size_t)blockIdx.x * 256 + threadIdx.x; t < total;
       t += (size_t)gridDim.x * 256) {
    const size_t b = t / ((size_t)I * J);
    const int r = (int)(t % ((size_t)I * J));
    const int j = r / I, i = r % I;
    out[t] = (bf16)in[b * I * J + (size_t)i * J + j];
  }
}
// out[j][i] (j < Jpad, zero-padded) = in[i][j];  in is (I x J) f32
__global__ void k_tr_pad(const float* __restrict__ in, bf16* __restrict__ out,
                         int I, int J, int Jpad) {
  const size_t total = (size_t)Jpad * I;
  for (size_t t = (size_t)blockIdx.x * 256 + threadIdx.x; t < total;
       t += (size_t)gridDim.x * 256) {
    const int j = (int)(t / I), i = (int)(t % I);
    out[t] = (j < J) ? (bf16)in[(size_t)i * J + j] : (bf16)0.f;
  }
}
// mh_w (32,256,3,3) -> out[tap][n][k] = mh_w[n][k][tap]
__global__ void k_trmh(const float* __restrict__ in, bf16* __restrict__ out) {
  const int t = blockIdx.x * 256 + threadIdx.x;
  if (t < 9 * 32 * 256) {
    const int tap = t / (32 * 256);
    const int r = t % (32 * 256);
    const int n = r / 256, k = r % 256;
    out[t] = (bf16)in[n * 2304 + k * 9 + tap];
  }
}

// ---------------------------------------------------------------------------
// K1: level-5 lateral. ffb(b,pix,256) = x5t(b,pix,512) @ latWb^T(256n x 512k)
// grid (4,16), block 512 (16 waves = 16 N tiles), K=512.
// ---------------------------------------------------------------------------
__global__ void k_lat5(const bf16* __restrict__ x5t, const bf16* __restrict__ latWb,
                       const float* __restrict__ lat_b, bf16* __restrict__ ffb) {
  const int b = blockIdx.x, mt = blockIdx.y;
  const int lane = threadIdx.x & 31, wave = threadIdx.x >> 5;
  const bf16* A = x5t + (size_t)(b * 256 + mt * 16) * 512;
  const int n0 = wave * 16;
  v8f acc = {};
#pragma unroll
  for (int kk = 0; kk < 512; kk += 32)
    acc = wmma_bf16(loadA(A, 512, lane, kk), loadBT(latWb, 512, lane, n0, kk), acc);
  const int n = n0 + (lane & 15);
  const float bias = lat_b[n];
#pragma unroll
  for (int r = 0; r < 8; ++r) {
    const int m = mt * 16 + r + ((lane < 16) ? 0 : 8);
    ffb[((size_t)b * 256 + m) * 256 + n] = (bf16)(acc[r] + bias);
  }
}

// ---------------------------------------------------------------------------
// K2: fused MLP: (Linear -> LN -> SiLU)*4 -> Linear(dout) [opt sigmoid].
// One block = 16 tokens; activations LDS-resident; input staged via
// GLOBAL_LOAD_ASYNC_TO_LDS_B128. Wt: packed bf16 [layer][n][k]; WoutT: [nPad][k].
// ---------------------------------------------------------------------------
__global__ void k_mlp(const bf16* __restrict__ X0,
                      const bf16* __restrict__ Wt, const float* __restrict__ Bb,
                      const float* __restrict__ G, const float* __restrict__ Beta,
                      const bf16* __restrict__ WoutT, const float* __restrict__ Bout,
                      int dout, int mode, float* __restrict__ out) {
  __shared__ __align__(16) bf16 Xs[16 * NCH];
  __shared__ float Ys[16 * NCH];
  __shared__ float mu[16], rs[16];
  const int tb = blockIdx.x;
  const int tid = threadIdx.x, lane = tid & 31, wave = tid >> 5;

  // async stage 16x256 bf16 (8KB) -> LDS: 512 lanes x 16B
  {
    const char* src = (const char*)X0 + (size_t)tb * 16 * NCH * sizeof(bf16) + tid * 16;
    const uint32_t dst = (uint32_t)(uintptr_t)&Xs[0] + tid * 16;
    async_copy16(src, dst);
    wait_async0();
  }
  __syncthreads();

  for (int layer = 0; layer < 4; ++layer) {
    const bf16* Wl = Wt + (size_t)layer * NCH * NCH;
    if (layer < 3) __builtin_prefetch(Wt + (size_t)(layer + 1) * NCH * NCH, 0, 1);
    const int n0 = wave * 16;
    v8f acc = {};
#pragma unroll
    for (int kk = 0; kk < NCH; kk += 32)
      acc = wmma_bf16(loadA(Xs, NCH, lane, kk), loadBT(Wl, NCH, lane, n0, kk), acc);
    const int n = n0 + (lane & 15);
    const float bias = Bb[layer * NCH + n];
#pragma unroll
    for (int r = 0; r < 8; ++r)
      Ys[(r + ((lane < 16) ? 0 : 8)) * NCH + n] = acc[r] + bias;
    __syncthreads();
    // LayerNorm: wave w reduces token-row w (wave32 xor tree)
    {
      float s = 0.f, s2 = 0.f;
      for (int c = lane; c < NCH; c += 32) {
        const float v = Ys[wave * NCH + c];
        s += v; s2 += v * v;
      }
#pragma unroll
      for (int off = 16; off > 0; off >>= 1) {
        s += __shfl_xor(s, off, 32);
        s2 += __shfl_xor(s2, off, 32);
      }
      if (lane == 0) {
        const float m_ = s * (1.f / NCH);
        mu[wave] = m_;
        rs[wave] = rsqrtf(s2 * (1.f / NCH) - m_ * m_ + 1e-5f);
      }
    }
    __syncthreads();
    for (int e = tid; e < 16 * NCH; e += 512) {
      const int row = e >> 8, c = e & 255;
      const float v =
          (Ys[e] - mu[row]) * rs[row] * G[layer * NCH + c] + Beta[layer * NCH + c];
      Xs[e] = (bf16)silu_f(v);
    }
    __syncthreads();
  }

  for (int nt = wave; nt * 16 < dout; nt += 16) {
    const int n0 = nt * 16;
    v8f acc = {};
#pragma unroll
    for (int kk = 0; kk < NCH; kk += 32)
      acc = wmma_bf16(loadA(Xs, NCH, lane, kk), loadBT(WoutT, NCH, lane, n0, kk), acc);
    const int n = n0 + (lane & 15);
    if (n < dout) {
      const float bias = Bout[n];
#pragma unroll
      for (int r = 0; r < 8; ++r) {
        const int m = r + ((lane < 16) ? 0 : 8);
        float v = acc[r] + bias;
        if (mode == 1) v = sigmoid_f(v);
        out[((size_t)tb * 16 + m) * dout + n] = v;
      }
    }
  }
}

// ---------------------------------------------------------------------------
// K3: top-100 (bitonic in LDS), scores, counts, instance gather, grid offsets.
// ---------------------------------------------------------------------------
__global__ void k_topk(const float* __restrict__ logits, const bf16* __restrict__ ffb,
                       float* __restrict__ num_out, float* __restrict__ scores_out,
                       bf16* __restrict__ instf, float* __restrict__ moff) {
  __shared__ float sv[256];
  __shared__ int si[256];
  __shared__ int cnt;
  const int b = blockIdx.x, tid = threadIdx.x;
  sv[tid] = logits[b * 256 + tid];
  si[tid] = tid;
  if (tid == 0) cnt = 0;
  __syncthreads();
  for (int k = 2; k <= 256; k <<= 1)
    for (int j = k >> 1; j > 0; j >>= 1) {
      const int p = tid ^ j;
      if (p > tid) {
        const bool descend = ((tid & k) == 0);
        const float v0 = sv[tid], v1 = sv[p];
        const int i0 = si[tid], i1 = si[p];
        const bool firstBigger = (v0 > v1) || (v0 == v1 && i0 < i1);
        if (descend ? !firstBigger : firstBigger) {
          sv[tid] = v1; sv[p] = v0; si[tid] = i1; si[p] = i0;
        }
      }
      __syncthreads();
    }
  if (tid < MAXI) {
    const float sc = sigmoid_f(sv[tid]);
    scores_out[b * MAXI + tid] = sc;
    if (sc > 0.5f) atomicAdd(&cnt, 1);
    const int idx = si[tid];
    moff[(b * MAXI + tid) * 2 + 0] = ((idx & 15) + 0.5f) / 16.f;  // x
    moff[(b * MAXI + tid) * 2 + 1] = ((idx >> 4) + 0.5f) / 16.f;  // y
  }
  __syncthreads();
  if (tid == 0) num_out[b] = (float)cnt;
  for (int e = tid; e < MAXI * NCH; e += 256) {
    const int inst = e >> 8, c = e & 255;
    instf[((size_t)b * MAXI + inst) * NCH + c] =
        ffb[((size_t)b * 256 + si[inst]) * NCH + c];
  }
}

// ---------------------------------------------------------------------------
// K4: mask lateral. mfl(b,pix,256) = x3t(b,pix,256) @ mlWb^T. grid (4,256), 512 thr.
// ---------------------------------------------------------------------------
__global__ void k_mlat(const bf16* __restrict__ x3t, const bf16* __restrict__ mlWb,
                       const float* __restrict__ ml_b, bf16* __restrict__ mfl) {
  const int b = blockIdx.x, mt = blockIdx.y;
  const int lane = threadIdx.x & 31, wave = threadIdx.x >> 5;
  const bf16* A = x3t + (size_t)(b * 4096 + mt * 16) * 256;
  const int n0 = wave * 16;
  v8f acc = {};
#pragma unroll
  for (int kk = 0; kk < 256; kk += 32)
    acc = wmma_bf16(loadA(A, 256, lane, kk), loadBT(mlWb, 256, lane, n0, kk), acc);
  const int n = n0 + (lane & 15);
  const float bias = ml_b[n];
#pragma unroll
  for (int r = 0; r < 8; ++r) {
    const int m = mt * 16 + r + ((lane < 16) ? 0 : 8);
    mfl[((size_t)b * 4096 + m) * 256 + n] = (bf16)(acc[r] + bias);
  }
}

// ---------------------------------------------------------------------------
// K5: 3x3 mask head as 9 shifted WMMA GEMMs + SiLU. grid (4,256), block 64.
// Per-lane boundary predication keeps the whole 16B K-run vectorized.
// ---------------------------------------------------------------------------
__global__ void k_mhead(const bf16* __restrict__ mfl, const bf16* __restrict__ mhWt,
                        const float* __restrict__ mh_b, bf16* __restrict__ mf) {
  const int b = blockIdx.x, mt = blockIdx.y;
  const int lane = threadIdx.x & 31, wave = threadIdx.x >> 5;
  const int n0 = wave * 16;
  const bf16* In = mfl + (size_t)b * 4096 * 256;
  const int m = lane & 15;
  const int koff = (lane < 16) ? 0 : 8;
  const int pix = mt * 16 + m;
  v8f acc = {};
#pragma unroll
  for (int tap = 0; tap < 9; ++tap) {
    const int dy = tap / 3 - 1, dx = tap % 3 - 1;
    const int y = (pix >> 6) + dy, x = (pix & 63) + dx;
    const bool inb = ((unsigned)y < 64u) && ((unsigned)x < 64u);
    const bf16* row = In + (size_t)(inb ? (y * 64 + x) : 0) * 256;
#pragma unroll
    for (int kk = 0; kk < 256; kk += 32) {
      const bf16* p = row + kk + koff;
      v8bf z = {};
      v8bf lo = inb ? *(const v8bf*)p : z;
      v8bf hi = inb ? *(const v8bf*)(p + 16) : z;
      v16bf af = __builtin_shufflevector(lo, hi, 0, 1, 2, 3, 4, 5, 6, 7,
                                         8, 9, 10, 11, 12, 13, 14, 15);
      acc = wmma_bf16(af, loadBT(mhWt + (size_t)tap * 32 * 256, 256, lane, n0, kk),
                      acc);
    }
  }
  const int n = n0 + (lane & 15);
  const float bias = mh_b[n];
#pragma unroll
  for (int r = 0; r < 8; ++r) {
    const int mm = mt * 16 + r + ((lane < 16) ? 0 : 8);
    mf[((size_t)b * 4096 + mm) * C_CH + n] = (bf16)silu_f(acc[r] + bias);
  }
}

// ---------------------------------------------------------------------------
// K6: per-instance dynamic conv chain + keypoint argmax. grid (4,100), block 256.
// Coordinate columns of layer-1 folded in as 2 VALU FMAs (exact, no padded WMMA).
// ---------------------------------------------------------------------------
__global__ void k_dyn(const bf16* __restrict__ mf, const float* __restrict__ dw,
                      const float* __restrict__ moff, float* __restrict__ kpts) {
  __shared__ __align__(16) bf16 w1t[C_CH][C_CH];  // [n][k]
  __shared__ __align__(16) bf16 w2t[C_CH][C_CH];  // [n][k]
  __shared__ __align__(16) bf16 w3t[C_CH][C_CH];  // [n][k], n 17..31 zero
  __shared__ float cgx[C_CH], cgy[C_CH], b1s[C_CH], b2s[C_CH], b3s[C_CH];
  __shared__ __align__(16) bf16 hs[8][16][C_CH];
  __shared__ float cval[8][KPN][2];
  __shared__ int cidx[8][KPN][2];

  const int b = blockIdx.x, inst = blockIdx.y;
  const int tid = threadIdx.x, lane = tid & 31, wave = tid >> 5;
  const float* dwp = dw + ((size_t)b * MAXI + inst) * KPAR;

  for (int e = tid; e < C_CH * C_CH; e += 256) {
    const int n = e >> 5, k = e & 31;
    w1t[n][k] = (bf16)dwp[k * C_CH + n];
    w2t[n][k] = (bf16)dwp[1120 + k * C_CH + n];
    w3t[n][k] = (n < KPN) ? (bf16)dwp[2176 + k * KPN + n] : (bf16)0.f;
  }
  if (tid < C_CH) {
    cgx[tid] = dwp[1024 + tid];   // w1 row 32 (gx)
    cgy[tid] = dwp[1056 + tid];   // w1 row 33 (gy)
    b1s[tid] = dwp[1088 + tid];
    b2s[tid] = dwp[2144 + tid];
    b3s[tid] = (tid < KPN) ? dwp[2720 + tid] : 0.f;
  }
  const float offx = moff[((size_t)b * MAXI + inst) * 2 + 0];
  const float offy = moff[((size_t)b * MAXI + inst) * 2 + 1];
  __syncthreads();

  const bf16* MF = mf + (size_t)b * 4096 * C_CH;
  const int hi = (lane < 16) ? 0 : 8;
  float best0 = -1e30f; int bi0 = 0;   // keypoint = lane&15
  float best1 = -1e30f; int bi1 = 0;   // keypoint 16 on lanes with lane&15==0

  for (int t = wave; t < 256; t += 8) {
    const int pixbase = t * 16;
    // layer 1: mf(16x32) @ w1 + coord FMAs
    const v16bf a1 = loadA(MF + (size_t)pixbase * C_CH, C_CH, lane, 0);
#pragma unroll
    for (int nt = 0; nt < 2; ++nt) {
      v8f acc = {};
      acc = wmma_bf16(a1, loadBT(&w1t[0][0], C_CH, lane, nt * 16, 0), acc);
      const int col = nt * 16 + (lane & 15);
      const float wgx = cgx[col], wgy = cgy[col], bias = b1s[col];
#pragma unroll
      for (int r = 0; r < 8; ++r) {
        const int pix = pixbase + r + hi;
        const float relx = ((pix & 63) + 0.5f) * (1.f / 64.f) - offx;
        const float rely = ((pix >> 6) + 0.5f) * (1.f / 64.f) - offy;
        const float v = acc[r] + relx * wgx + rely * wgy + bias;
        hs[wave][r + hi][col] = (bf16)silu_f(v);
      }
    }
    // layer 2: h1(16x32) @ w2
    {
      const v16bf a2 = loadA(&hs[wave][0][0], C_CH, lane, 0);
      v8f acc0 = {}, acc1 = {};
      acc0 = wmma_bf16(a2, loadBT(&w2t[0][0], C_CH, lane, 0, 0), acc0);
      acc1 = wmma_bf16(a2, loadBT(&w2t[0][0], C_CH, lane, 16, 0), acc1);
#pragma unroll
      for (int nt = 0; nt < 2; ++nt) {
        const v8f acc = nt ? acc1 : acc0;
        const int col = nt * 16 + (lane & 15);
        const float bias = b2s[col];
#pragma unroll
        for (int r = 0; r < 8; ++r)
          hs[wave][r + hi][col] = (bf16)silu_f(acc[r] + bias);
      }
    }
    // layer 3: h2(16x32) @ w3 (N padded 17->32) + argmax update
    {
      const v16bf a3 = loadA(&hs[wave][0][0], C_CH, lane, 0);
#pragma unroll
      for (int nt = 0; nt < 2; ++nt) {
        v8f acc = {};
        acc = wmma_bf16(a3, loadBT(&w3t[0][0], C_CH, lane, nt * 16, 0), acc);
        const int col = nt * 16 + (lane & 15);
        if (col < KPN) {
          const float bias = b3s[col];
#pragma unroll
          for (int r = 0; r < 8; ++r) {
            const float v = acc[r] + bias;
            const int pix = pixbase + r + hi;
            if (nt == 0) { if (v > best0) { best0 = v; bi0 = pix; } }
            else         { if (v > best1) { best1 = v; bi1 = pix; } }
          }
        }
      }
    }
  }

  const int half = lane >> 4;
  cval[wave][lane & 15][half] = best0;
  cidx[wave][lane & 15][half] = bi0;
  if ((lane & 15) == 0) { cval[wave][16][half] = best1; cidx[wave][16][half] = bi1; }
  __syncthreads();

  if (tid < KPN) {
    float bv = -1e30f; int bidx = 0;
    for (int w = 0; w < 8; ++w)
      for (int h = 0; h < 2; ++h) {
        const float v = cval[w][tid][h]; const int ix = cidx[w][tid][h];
        if (v > bv || (v == bv && ix < bidx)) { bv = v; bidx = ix; }
      }
    float* kp = kpts + (((size_t)b * MAXI + inst) * KPN + tid) * 2;
    kp[0] = ((bidx & 63) + 0.5f) / 64.f * 512.f;  // x
    kp[1] = ((bidx >> 6) + 0.5f) / 64.f * 512.f;  // y
  }
}

// ---------------------------------------------------------------------------
// Launcher
// ---------------------------------------------------------------------------
extern "C" void kernel_launch(void* const* d_in, const int* in_sizes, int n_in,
                              void* d_out, int out_size, void* d_ws, size_t ws_size,
                              hipStream_t stream) {
  (void)in_sizes; (void)n_in; (void)out_size; (void)ws_size;
  const float* x3    = (const float*)d_in[1];
  const float* x5    = (const float*)d_in[2];
  const float* lat_w = (const float*)d_in[3];
  const float* lat_b = (const float*)d_in[4];
  const float* ml_w  = (const float*)d_in[5];
  const float* ml_b  = (const float*)d_in[6];
  const float* mh_w  = (const float*)d_in[7];
  const float* mh_b  = (const float*)d_in[8];
  const float* loc_p[6]; const float* pre_p[6]; const float* ker_p[6];
  for (int j = 0; j < 6; ++j) {
    loc_p[j] = (const float*)d_in[9 + j];
    pre_p[j] = (const float*)d_in[15 + j];
    ker_p[j] = (const float*)d_in[21 + j];
  }

  char* ws = (char*)d_ws;
  size_t off = 0;
  auto alloc = [&](size_t bytes) {
    char* p = ws + off; off += (bytes + 255) & ~size_t(255); return p;
  };
  bf16*  ffb    = (bf16*) alloc((size_t)4 * 256 * NCH * sizeof(bf16));
  float* logits = (float*)alloc(4 * 256 * sizeof(float));
  bf16*  instf  = (bf16*) alloc((size_t)4 * MAXI * NCH * sizeof(bf16));
  float* moff   = (float*)alloc(4 * MAXI * 2 * sizeof(float));
  float* dwv    = (float*)alloc((size_t)4 * MAXI * KPAR * sizeof(float));
  bf16*  mfl    = (bf16*) alloc((size_t)4 * 4096 * NCH * sizeof(bf16));
  bf16*  mf     = (bf16*) alloc((size_t)4 * 4096 * C_CH * sizeof(bf16));
  bf16*  x5t    = (bf16*) alloc((size_t)4 * 256 * 512 * sizeof(bf16));
  bf16*  x3t    = (bf16*) alloc((size_t)4 * 4096 * 256 * sizeof(bf16));
  bf16*  latWb  = (bf16*) alloc((size_t)256 * 512 * sizeof(bf16));
  bf16*  mlWb   = (bf16*) alloc((size_t)256 * 256 * sizeof(bf16));
  bf16*  mhWt   = (bf16*) alloc((size_t)9 * 32 * 256 * sizeof(bf16));
  bf16*  locWt  = (bf16*) alloc((size_t)4 * NCH * NCH * sizeof(bf16));
  bf16*  preWt  = (bf16*) alloc((size_t)4 * NCH * NCH * sizeof(bf16));
  bf16*  kerWt  = (bf16*) alloc((size_t)4 * NCH * NCH * sizeof(bf16));
  bf16*  locWoT = (bf16*) alloc((size_t)16 * NCH * sizeof(bf16));     // pad 1->16
  bf16*  preWoT = (bf16*) alloc((size_t)32 * NCH * sizeof(bf16));     // pad 17->32
  bf16*  kerWoT = (bf16*) alloc((size_t)2752 * NCH * sizeof(bf16));   // pad 2737->2752

  float* out_num  = (float*)d_out;
  float* out_sc   = out_num + 4;
  float* out_pre  = out_sc + 4 * MAXI;
  float* out_kpts = out_pre + 4 * MAXI * KPN;

  // ---- pack phase ----
  k_tr<<<2048, 256, 0, stream>>>(x5, x5t, 4, 512, 256);                 // (4,256,512)
  k_tr<<<16384, 256, 0, stream>>>(x3, x3t, 4, 256, 4096);               // (4,4096,256)
  k_cvt<<<512, 256, 0, stream>>>(lat_w, latWb, 256 * 512);              // already [n][k]
  k_cvt<<<256, 256, 0, stream>>>(ml_w, mlWb, 256 * 256);                // already [n][k]
  k_trmh<<<(9 * 32 * 256 + 255) / 256, 256, 0, stream>>>(mh_w, mhWt);
  k_tr<<<1024, 256, 0, stream>>>(loc_p[0], locWt, 4, NCH, NCH);
  k_tr<<<1024, 256, 0, stream>>>(pre_p[0], preWt, 4, NCH, NCH);
  k_tr<<<1024, 256, 0, stream>>>(ker_p[0], kerWt, 4, NCH, NCH);
  k_tr_pad<<<16, 256, 0, stream>>>(loc_p[4], locWoT, NCH, 1, 16);
  k_tr_pad<<<32, 256, 0, stream>>>(pre_p[4], preWoT, NCH, KPN, 32);
  k_tr_pad<<<2752, 256, 0, stream>>>(ker_p[4], kerWoT, NCH, KPAR, 2752);

  // ---- main pipeline ----
  k_lat5<<<dim3(4, 16), 512, 0, stream>>>(x5t, latWb, lat_b, ffb);
  k_mlp<<<64, 512, 0, stream>>>(ffb, locWt, loc_p[1], loc_p[2], loc_p[3],
                                locWoT, loc_p[5], 1, 0, logits);
  k_topk<<<4, 256, 0, stream>>>(logits, ffb, out_num, out_sc, instf, moff);
  k_mlp<<<25, 512, 0, stream>>>(instf, preWt, pre_p[1], pre_p[2], pre_p[3],
                                preWoT, pre_p[5], KPN, 1, out_pre);
  k_mlp<<<25, 512, 0, stream>>>(instf, kerWt, ker_p[1], ker_p[2], ker_p[3],
                                kerWoT, ker_p[5], KPAR, 0, dwv);
  k_mlat<<<dim3(4, 256), 512, 0, stream>>>(x3t, mlWb, ml_b, mfl);
  k_mhead<<<dim3(4, 256), 64, 0, stream>>>(mfl, mhWt, mh_b, mf);
  k_dyn<<<dim3(4, MAXI), 256, 0, stream>>>(mf, dwv, moff, out_kpts);
}
